// WavLMAttention_59588376265097
// MI455X (gfx1250) — matrix operand order repaired and checked by reading, not compiled, not run on hardware
//
#include <hip/hip_runtime.h>
#include <hip/hip_bf16.h>
#include <math.h>

// ---------------------------------------------------------------------------
// WavLM gated-relative-bias attention for MI455X (gfx1250, wave32, WMMA).
// B=4, N=1024, E=1024, H=16, D=64.
// bf16 WMMA for all matmuls; flash attention with double-buffered
// GLOBAL_LOAD_ASYNC_TO_LDS_B128 K/V staging (ASYNCcnt-tracked DMA overlapped
// with WMMA compute); P re-layout C->A through per-wave LDS slice.
// ---------------------------------------------------------------------------

typedef __attribute__((ext_vector_type(16))) __bf16 v16bf;
typedef __attribute__((ext_vector_type(8)))  float  v8f;

#define BB   4
#define NSEQ 1024
#define EE   1024
#define HH   16
#define DD   64

static __device__ __forceinline__ __bf16 f2bf(float f) {
  union { float f; unsigned u; } in; in.f = f;
  unsigned u = in.u;
  u += 0x7FFFu + ((u >> 16) & 1u);          // round-to-nearest-even
  union { unsigned short s; __bf16 b; } out;
  out.s = (unsigned short)(u >> 16);
  return out.b;
}

// Issue a 16-byte async DMA from global memory directly into LDS.
// VDST carries the LDS byte address (generic shared ptr low 32 bits == LDS
// offset per the flat-address LDS aperture decode); VADDR is the 64-bit
// global address; tracked with ASYNCcnt.
static __device__ __forceinline__ void async_load_b128(const void* gaddr,
                                                       void* lds) {
  unsigned ldst = (unsigned)(size_t)lds;
  unsigned long long ga = (unsigned long long)(size_t)gaddr;
  asm volatile("global_load_async_to_lds_b128 %0, %1, off"
               :: "v"(ldst), "v"(ga) : "memory");
}

// ---------------------------------------------------------------------------
// fp32 -> bf16 pack
// ---------------------------------------------------------------------------
__global__ void f32_to_bf16_kernel(const float* __restrict__ src,
                                   __bf16* __restrict__ dst, int n) {
  int i = blockIdx.x * blockDim.x + threadIdx.x;
  if (i < n) dst[i] = f2bf(src[i]);
}

// ---------------------------------------------------------------------------
// GEMM: C[m][n] = sum_k A[m][k] * W[n][k] + bias[n]
// A: [4096 x 1024] bf16, W: [1024 x 1024] bf16 (row-major, so W^T multiply).
// Each wave computes one 16x16 C tile; 8 waves / block.
// ---------------------------------------------------------------------------
__global__ void __launch_bounds__(256)
gemm_qkv_kernel(const __bf16* __restrict__ A, const __bf16* __restrict__ W,
                const float* __restrict__ bias, __bf16* __restrict__ outBHND) {
  const int K = EE, TN = EE / 16;           // 64 col tiles
  const int lane = threadIdx.x & 31;
  const int wave = threadIdx.x >> 5;
  const int wid  = blockIdx.x * 8 + wave;
  const int tm = wid / TN, tn = wid % TN;
  const int row  = tm * 16 + (lane & 15);
  const int kh   = lane >> 4;
  const int coln = tn * 16 + (lane & 15);

  const __bf16* arow = A + (size_t)row  * K;
  const __bf16* wrow = W + (size_t)coln * K;
  v8f acc = {};
  for (int k0 = 0; k0 < K; k0 += 32) {
    __builtin_prefetch(arow + k0 + 128, 0, 1);   // global_prefetch_b8
    v16bf a, b;
#pragma unroll
    for (int j = 0; j < 8; ++j) {
      int ka = 16 * kh + 2 * j;                                // contiguous k layout
      a[2 * j]     = arow[k0 + ka];
      a[2 * j + 1] = arow[k0 + ka + 1];
      int kb = 16 * kh + 2 * j;                                // contiguous k layout (same as A)
      b[2 * j]     = wrow[k0 + kb];
      b[2 * j + 1] = wrow[k0 + kb + 1];
    }
    acc = __builtin_amdgcn_wmma_f32_16x16x32_bf16(false, a, false, b,
                                                  (short)0, acc, false, false);
  }
  const float bs = bias[coln];
  const int h = coln / DD, d = coln % DD;
#pragma unroll
  for (int r = 0; r < 8; ++r) {
    int m  = tm * 16 + r + 8 * kh;            // C layout row
    int bi = m >> 10, i = m & (NSEQ - 1);
    outBHND[(((size_t)bi * HH + h) * NSEQ + i) * DD + d] = f2bf(acc[r] + bs);
  }
}

__global__ void __launch_bounds__(256)
gemm_out_kernel(const __bf16* __restrict__ A, const __bf16* __restrict__ W,
                const float* __restrict__ bias, float* __restrict__ out) {
  const int K = EE, TN = EE / 16;
  const int lane = threadIdx.x & 31;
  const int wave = threadIdx.x >> 5;
  const int wid  = blockIdx.x * 8 + wave;
  const int tm = wid / TN, tn = wid % TN;
  const int row  = tm * 16 + (lane & 15);
  const int kh   = lane >> 4;
  const int coln = tn * 16 + (lane & 15);

  const __bf16* arow = A + (size_t)row  * K;
  const __bf16* wrow = W + (size_t)coln * K;
  v8f acc = {};
  for (int k0 = 0; k0 < K; k0 += 32) {
    __builtin_prefetch(arow + k0 + 128, 0, 1);
    v16bf a, b;
#pragma unroll
    for (int j = 0; j < 8; ++j) {
      int ka = 16 * kh + 2 * j;
      a[2 * j]     = arow[k0 + ka];
      a[2 * j + 1] = arow[k0 + ka + 1];
      int kb = 16 * kh + 2 * j;
      b[2 * j]     = wrow[k0 + kb];
      b[2 * j + 1] = wrow[k0 + kb + 1];
    }
    acc = __builtin_amdgcn_wmma_f32_16x16x32_bf16(false, a, false, b,
                                                  (short)0, acc, false, false);
  }
  const float bs = bias[coln];
#pragma unroll
  for (int r = 0; r < 8; ++r) {
    int m = tm * 16 + r + 8 * kh;
    out[(size_t)m * EE + coln] = acc[r] + bs;
  }
}

// ---------------------------------------------------------------------------
// Relative-position bucket map (head-independent): bucket[i*N + j]
// ---------------------------------------------------------------------------
__global__ void bucket_kernel(int* __restrict__ bucket) {
  int idx = blockIdx.x * blockDim.x + threadIdx.x;
  if (idx >= NSEQ * NSEQ) return;
  int i = idx >> 10, j = idx & (NSEQ - 1);
  int rel = j - i;
  const int nb = 160, max_exact = 80;
  int base = (rel > 0) ? nb : 0;
  int a = abs(rel);
  int v;
  if (a < max_exact) {
    v = a;
  } else {
    float large = logf((float)a * (1.0f / 80.0f)) *
                  ((float)(nb - max_exact) / logf(800.0f / 80.0f));
    v = (int)(80.0f + large);
    if (v > nb - 1) v = nb - 1;
  }
  bucket[idx] = base + v;
}

// ---------------------------------------------------------------------------
// GRU-style gate: one thread per (b,h,i) -> gate_out scalar
// ---------------------------------------------------------------------------
__global__ void gate_kernel(const float* __restrict__ hidden,
                            const float* __restrict__ gruW,
                            const float* __restrict__ gruB,
                            const float* __restrict__ gruC,
                            float* __restrict__ gate) {
  int idx = blockIdx.x * blockDim.x + threadIdx.x;
  if (idx >= BB * HH * NSEQ) return;
  int i = idx & (NSEQ - 1);
  int h = (idx >> 10) & (HH - 1);
  int b = idx >> 14;
  const float* x = hidden + ((size_t)b * NSEQ + i) * EE + h * DD;
  float g0 = 0.f, g1 = 0.f;
#pragma unroll
  for (int c = 0; c < 8; ++c) {
    float p = gruB[c];
    for (int d = 0; d < DD; ++d) p += x[d] * gruW[c * DD + d];
    if (c < 4) g0 += p; else g1 += p;
  }
  float s0 = 1.0f / (1.0f + __expf(-g0));
  float s1 = 1.0f / (1.0f + __expf(-g1));
  gate[idx] = s0 * (s1 * gruC[h] - 1.0f) + 2.0f;
}

// ---------------------------------------------------------------------------
// Flash attention. grid = (B*H, N/128); 256 threads = 8 waves; each wave owns
// 16 query rows. K/V 32x64 bf16 tiles double-buffered in LDS via
// global_load_async_to_lds_b128 (one B128 per thread per tile), so the next
// tile's DMA overlaps the current tile's WMMAs. s_wait_asynccnt 2 retires the
// current buffer while the next buffer's 2 loads remain in flight.
// ---------------------------------------------------------------------------
__global__ void __launch_bounds__(256)
flash_attn_kernel(const __bf16* __restrict__ Q, const __bf16* __restrict__ Km,
                  const __bf16* __restrict__ V, const int* __restrict__ bucket,
                  const float* __restrict__ gate,
                  const float* __restrict__ rel_embed,
                  __bf16* __restrict__ O) {
  __shared__ float  relc[320];
  __shared__ alignas(16) __bf16 Ks[2][32][DD];
  __shared__ alignas(16) __bf16 Vs[2][32][DD];
  __shared__ alignas(16) __bf16 Ps[8][16][32];

  const int bh = blockIdx.x;
  const int h = bh & (HH - 1), b = bh >> 4;
  const int wave = threadIdx.x >> 5, lane = threadIdx.x & 31;
  const int i0 = blockIdx.y * 128 + wave * 16;
  const int arow = lane & 15, kh = lane >> 4, jc = lane & 15;

  for (int t = threadIdx.x; t < 320; t += 256) relc[t] = rel_embed[t * HH + h];

  const __bf16* Qb = Q + (size_t)bh * NSEQ * DD;
  const __bf16* Kb = Km + (size_t)bh * NSEQ * DD;
  const __bf16* Vb = V + (size_t)bh * NSEQ * DD;

  // Per-thread async-DMA chunk: 16B of the 32x64 tile (256 threads cover it).
  const int dmar = threadIdx.x >> 3;        // tile row 0..31
  const int dmac = (threadIdx.x & 7) * 8;   // bf16 column 0,8,...,56

  // Preload the wave's 16x64 Q tile as two 16x32 A-layout tiles.
  v16bf qa0, qa1;
  const __bf16* qrow = Qb + (size_t)(i0 + arow) * DD;
#pragma unroll
  for (int j = 0; j < 8; ++j) {
    int ka = 16 * kh + 2 * j;
    qa0[2 * j] = qrow[ka];      qa0[2 * j + 1] = qrow[ka + 1];
    qa1[2 * j] = qrow[32 + ka]; qa1[2 * j + 1] = qrow[32 + ka + 1];
  }

  float gate_r[8], m_r[8], l_r[8];
#pragma unroll
  for (int r = 0; r < 8; ++r) {
    int i = i0 + r + 8 * kh;
    gate_r[r] = gate[(size_t)bh * NSEQ + i];
    m_r[r] = -1e30f;
    l_r[r] = 0.0f;
  }
  v8f o[4] = {};
  const float scale = 0.125f;  // 1/sqrt(64)

  // Kick off DMA for the first K/V tile.
  async_load_b128(Kb + (size_t)dmar * DD + dmac, &Ks[0][dmar][dmac]);
  async_load_b128(Vb + (size_t)dmar * DD + dmac, &Vs[0][dmar][dmac]);

  for (int kb = 0; kb < NSEQ; kb += 32) {
    const int cur = (kb >> 5) & 1;
    if (kb + 32 < NSEQ) {
      // Prefetch next tile into the other buffer, then wait for current tile
      // (<=2 outstanding == only the just-issued pair may remain).
      async_load_b128(Kb + (size_t)(kb + 32 + dmar) * DD + dmac,
                      &Ks[cur ^ 1][dmar][dmac]);
      async_load_b128(Vb + (size_t)(kb + 32 + dmar) * DD + dmac,
                      &Vs[cur ^ 1][dmar][dmac]);
      asm volatile("s_wait_asynccnt 2" ::: "memory");
    } else {
      asm volatile("s_wait_asynccnt 0" ::: "memory");
    }
    __syncthreads();

    // S = Q K^T for key cols [kb, kb+32): two 16x16 C tiles, K-dim = 64.
    v8f s0 = {}, s1 = {};
    {
      v16bf bq;
#pragma unroll
      for (int t = 0; t < 8; ++t) {
        int dd = 16 * kh + 2 * t;
        bq[2 * t] = Ks[cur][jc][dd]; bq[2 * t + 1] = Ks[cur][jc][dd + 1];
      }
      s0 = __builtin_amdgcn_wmma_f32_16x16x32_bf16(false, qa0, false, bq,
                                                   (short)0, s0, false, false);
#pragma unroll
      for (int t = 0; t < 8; ++t) {
        int dd = 32 + 16 * kh + 2 * t;
        bq[2 * t] = Ks[cur][jc][dd]; bq[2 * t + 1] = Ks[cur][jc][dd + 1];
      }
      s0 = __builtin_amdgcn_wmma_f32_16x16x32_bf16(false, qa1, false, bq,
                                                   (short)0, s0, false, false);
#pragma unroll
      for (int t = 0; t < 8; ++t) {
        int dd = 16 * kh + 2 * t;
        bq[2 * t]     = Ks[cur][16 + jc][dd];
        bq[2 * t + 1] = Ks[cur][16 + jc][dd + 1];
      }
      s1 = __builtin_amdgcn_wmma_f32_16x16x32_bf16(false, qa0, false, bq,
                                                   (short)0, s1, false, false);
#pragma unroll
      for (int t = 0; t < 8; ++t) {
        int dd = 32 + 16 * kh + 2 * t;
        bq[2 * t]     = Ks[cur][16 + jc][dd];
        bq[2 * t + 1] = Ks[cur][16 + jc][dd + 1];
      }
      s1 = __builtin_amdgcn_wmma_f32_16x16x32_bf16(false, qa1, false, bq,
                                                   (short)0, s1, false, false);
    }

    // Gated bias + online softmax per row; write P (bf16) to per-wave LDS.
#pragma unroll
    for (int r = 0; r < 8; ++r) {
      int i = i0 + r + 8 * kh;
      const int* brow = bucket + (size_t)i * NSEQ + kb;
      float x0 = s0[r] * scale + gate_r[r] * relc[brow[jc]];
      float x1 = s1[r] * scale + gate_r[r] * relc[brow[16 + jc]];
      float mx = fmaxf(x0, x1);
#pragma unroll
      for (int d = 1; d < 16; d <<= 1) mx = fmaxf(mx, __shfl_xor(mx, d, 32));
      float nm    = fmaxf(m_r[r], mx);
      float alpha = __expf(m_r[r] - nm);
      float p0 = __expf(x0 - nm);
      float p1 = __expf(x1 - nm);
      float sum = p0 + p1;
#pragma unroll
      for (int d = 1; d < 16; d <<= 1) sum += __shfl_xor(sum, d, 32);
      l_r[r] = l_r[r] * alpha + sum;
      m_r[r] = nm;
      o[0][r] *= alpha; o[1][r] *= alpha; o[2][r] *= alpha; o[3][r] *= alpha;
      Ps[wave][r + 8 * kh][jc]      = f2bf(p0);
      Ps[wave][r + 8 * kh][16 + jc] = f2bf(p1);
    }

    // P (16x32, A-layout from our private LDS slice) @ V (32x64) -> O.
    v16bf pa;
#pragma unroll
    for (int j = 0; j < 8; ++j) {
      int ka = 16 * kh + 2 * j;
      pa[2 * j]     = Ps[wave][arow][ka];
      pa[2 * j + 1] = Ps[wave][arow][ka + 1];
    }
#pragma unroll
    for (int dt = 0; dt < 4; ++dt) {
      v16bf bv;
      int col = dt * 16 + jc;
#pragma unroll
      for (int t = 0; t < 8; ++t) {
        int jj = 16 * kh + 2 * t;
        bv[2 * t] = Vs[cur][jj][col]; bv[2 * t + 1] = Vs[cur][jj + 1][col];
      }
      o[dt] = __builtin_amdgcn_wmma_f32_16x16x32_bf16(false, pa, false, bv,
                                                      (short)0, o[dt], false, false);
    }
    // All waves must be done with buffer `cur` before anyone overwrites it
    // with the async issued at the top of the next iteration.
    __syncthreads();
  }

  // Normalize and store bf16 attention output in [B,N,H,D] (== [B,N,E]).
#pragma unroll
  for (int r = 0; r < 8; ++r) {
    int i = i0 + r + 8 * kh;
    float inv = 1.0f / l_r[r];
    size_t base = (((size_t)b * NSEQ + i) * HH + h) * DD;
#pragma unroll
    for (int dt = 0; dt < 4; ++dt)
      O[base + dt * 16 + jc] = f2bf(o[dt][r] * inv);
  }
}

// ---------------------------------------------------------------------------
// launcher
// ---------------------------------------------------------------------------
extern "C" void kernel_launch(void* const* d_in, const int* in_sizes, int n_in,
                              void* d_out, int out_size, void* d_ws, size_t ws_size,
                              hipStream_t stream) {
  (void)in_sizes; (void)n_in; (void)out_size; (void)ws_size;
  const float* hidden = (const float*)d_in[0];
  const float* Wq = (const float*)d_in[1];  const float* bq = (const float*)d_in[2];
  const float* Wk = (const float*)d_in[3];  const float* bk = (const float*)d_in[4];
  const float* Wv = (const float*)d_in[5];  const float* bv = (const float*)d_in[6];
  const float* Wo = (const float*)d_in[7];  const float* bo = (const float*)d_in[8];
  const float* gru_const = (const float*)d_in[9];
  const float* gru_W = (const float*)d_in[10];
  const float* gru_b = (const float*)d_in[11];
  const float* rel_embed = (const float*)d_in[12];

  const size_t nHid = (size_t)BB * NSEQ * EE;      // 4M
  const size_t nW   = (size_t)EE * EE;             // 1M
  const size_t nQKV = (size_t)BB * HH * NSEQ * DD; // 4M

  char* ws = (char*)d_ws;
  size_t off = 0;
  auto take = [&](size_t bytes) {
    void* p = ws + off;
    off += (bytes + 255) & ~(size_t)255;
    return p;
  };
  __bf16* hbf  = (__bf16*)take(nHid * 2);
  __bf16* wqbf = (__bf16*)take(nW * 2);
  __bf16* wkbf = (__bf16*)take(nW * 2);
  __bf16* wvbf = (__bf16*)take(nW * 2);
  __bf16* wobf = (__bf16*)take(nW * 2);
  __bf16* qbf  = (__bf16*)take(nQKV * 2);
  __bf16* kbf  = (__bf16*)take(nQKV * 2);
  __bf16* vbf  = (__bf16*)take(nQKV * 2);
  __bf16* abf  = (__bf16*)take(nQKV * 2);
  int*    bkt  = (int*)take((size_t)NSEQ * NSEQ * 4);
  float*  gat  = (float*)take((size_t)BB * HH * NSEQ * 4);

  // 1) pack fp32 -> bf16
  f32_to_bf16_kernel<<<(int)((nHid + 255) / 256), 256, 0, stream>>>(hidden, hbf, (int)nHid);
  f32_to_bf16_kernel<<<(int)((nW + 255) / 256), 256, 0, stream>>>(Wq, wqbf, (int)nW);
  f32_to_bf16_kernel<<<(int)((nW + 255) / 256), 256, 0, stream>>>(Wk, wkbf, (int)nW);
  f32_to_bf16_kernel<<<(int)((nW + 255) / 256), 256, 0, stream>>>(Wv, wvbf, (int)nW);
  f32_to_bf16_kernel<<<(int)((nW + 255) / 256), 256, 0, stream>>>(Wo, wobf, (int)nW);

  // 2) QKV projections -> head-split bf16 [B,H,N,D]
  const int gemmBlocks = (BB * NSEQ / 16) * (EE / 16) / 8;  // 2048
  gemm_qkv_kernel<<<gemmBlocks, 256, 0, stream>>>(hbf, wqbf, bq, qbf);
  gemm_qkv_kernel<<<gemmBlocks, 256, 0, stream>>>(hbf, wkbf, bk, kbf);
  gemm_qkv_kernel<<<gemmBlocks, 256, 0, stream>>>(hbf, wvbf, bv, vbf);

  // 3) position buckets + gates
  bucket_kernel<<<(NSEQ * NSEQ + 255) / 256, 256, 0, stream>>>(bkt);
  gate_kernel<<<(BB * HH * NSEQ + 255) / 256, 256, 0, stream>>>(
      hidden, gru_W, gru_b, gru_const, gat);

  // 4) flash attention -> bf16 [B,N,H,D] == [B,N,E]
  flash_attn_kernel<<<dim3(BB * HH, NSEQ / 128), 256, 0, stream>>>(
      qbf, kbf, vbf, bkt, gat, rel_embed, abf);

  // 5) output projection -> fp32 d_out [B,N,E]
  gemm_out_kernel<<<gemmBlocks, 256, 0, stream>>>(abf, wobf, bo, (float*)d_out);
}
